// CausalSelfAttention_43069932044288
// MI455X (gfx1250) — matrix operand order repaired and checked by reading, not compiled, not run on hardware
//
#include <hip/hip_runtime.h>
#include <hip/hip_bf16.h>
#include <stdint.h>

// ---------------------------------------------------------------------------
// Causal self-attention forward for MI455X (gfx1250).
// bf16 WMMA pipeline + Tensor Data Mover (TDM) staging of all LDS tiles.
//   B=4, T=2048, C=1024, H=16, dk=64
// ---------------------------------------------------------------------------

typedef __attribute__((ext_vector_type(8)))  float  v8f;
typedef __attribute__((ext_vector_type(8)))  __bf16 v8bf;
typedef __attribute__((ext_vector_type(16))) __bf16 v16bf;
typedef __attribute__((ext_vector_type(4)))  unsigned int v4u;
typedef __attribute__((ext_vector_type(8)))  int v8i;
typedef __attribute__((ext_vector_type(4)))  int v4i;

#define BATCH 4
#define TSEQ  2048
#define CDIM  1024
#define NH    16
#define DK    64
#define MTOT  (BATCH * TSEQ)                          // 8192
#define HEAD_ELEMS ((size_t)BATCH * NH * TSEQ * DK)   // elements in one of Q/K/V

#define LDSOFF(p) ((unsigned)(unsigned long long)(uintptr_t)(p))

// ---- fp32 -> bf16 (round-to-nearest-even) ---------------------------------
__device__ __forceinline__ unsigned short f2bf(float f) {
  unsigned u = __float_as_uint(f);
  unsigned r = 0x7FFFu + ((u >> 16) & 1u);
  return (unsigned short)((u + r) >> 16);
}

// ---------------------------------------------------------------------------
// Tensor Data Mover: 2D tile (16-bit elements) global -> LDS.
// D# packed per CDNA5 ISA §8.3/8.4:
//  g0: count=1 | lds_addr | global_addr[56:0] | type=2
//  g1: data_size=2B; tensor_dim0=row_stride; tensor_dim1=tile_y;
//      tile_dim0=tile_x; tile_dim1=tile_y; tensor_dim0_stride=row_stride
// Groups 2/3 zero (dims beyond 2D unused). Issued by one wave; tracked by
// TENSORcnt (S_WAIT_TENSORCNT), in-order per wave.
// ---------------------------------------------------------------------------
__device__ __forceinline__ void tdm_load_2d(unsigned lds_off, const void* gptr,
                                            unsigned tile_x, unsigned tile_y,
                                            unsigned row_stride) {
  const unsigned long long ga = (unsigned long long)(uintptr_t)gptr;
  v4u g0;
  g0[0] = 1u;                                    // count=1 (valid user D#)
  g0[1] = lds_off;                               // LDS byte address
  g0[2] = (unsigned)ga;                          // global_addr[31:0]
  g0[3] = (unsigned)((ga >> 32) & 0x01FFFFFFu)   // global_addr[56:32]
          | (2u << 30);                          // type=2 ("image")
  v8i g1;
  g1[0] = 0x00010000;                            // mask=0, data_size=1 (2B)
  g1[1] = (int)((row_stride & 0xFFFFu) << 16);   // tensor_dim0 lo (atomic=0)
  g1[2] = (int)((row_stride >> 16) | ((tile_y & 0xFFFFu) << 16)); // td0 hi | td1 lo
  g1[3] = (int)((tile_y >> 16) | (tile_x << 16));                 // td1 hi | tile_dim0
  g1[4] = (int)(tile_y & 0xFFFFu);               // tile_dim1 (tile_dim2=0)
  g1[5] = (int)row_stride;                       // tensor_dim0_stride[31:0]
  g1[6] = 0;                                     // stride hi | dim1_stride lo
  g1[7] = 0;
  v4i z4 = {0, 0, 0, 0};
  v8i z8 = {0, 0, 0, 0, 0, 0, 0, 0};
  // 6-arg form (clang-23 / therock-10.0 headers): groups 2,3 + extra group, cpol
  __builtin_amdgcn_tensor_load_to_lds(g0, g1, z4, z4, z8, 0);
}

// ---- WMMA fragment loaders (contiguous k order, same for A and B) -----------------
// A (16x32 bf16, row-major tile in LDS): lane holds row M=lane%16.
//   lanes 0-15 : elems 0-15 = K 0-15
//   lanes 16-31: elems 0-15 = K 16-31
__device__ __forceinline__ v16bf load_frag_a(const unsigned short* base,
                                             int pitch, int row, int kOff, int lane) {
  const int m = lane & 15, g = lane >> 4;
  const unsigned short* p = base + (size_t)(row + m) * pitch + kOff + g * 16;
  v8bf lo = *(const v8bf*)(p);
  v8bf hi = *(const v8bf*)(p + 8);
  v16bf out;
#pragma unroll
  for (int i = 0; i < 8; ++i) { out[i] = lo[i]; out[8 + i] = hi[i]; }
  return out;
}

// B (32x16 bf16): lane holds column N=lane%16; lanes 0-15 K=0..15,
// lanes 16-31 K=16..31.  Tile stored "N-major" in LDS: base[n][k].
__device__ __forceinline__ v16bf load_frag_b(const unsigned short* base,
                                             int pitch, int nRow, int kOff, int lane) {
  const int n = lane & 15, g = lane >> 4;
  const unsigned short* p = base + (size_t)(nRow + n) * pitch + kOff + g * 16;
  return *(const v16bf*)(p);
}

__device__ __forceinline__ v8f wmma_bf16(v16bf a, v16bf b, v8f c) {
  return __builtin_amdgcn_wmma_f32_16x16x32_bf16(false, a, false, b,
                                                 (short)0, c, false, false);
}

// ---------------------------------------------------------------------------
// Kernel 1a: fp32 -> bf16 bulk convert (vectorized x4)
// ---------------------------------------------------------------------------
__global__ __launch_bounds__(256) void cvt_bf16_kernel(const float* __restrict__ in,
                                                       unsigned short* __restrict__ out,
                                                       int n4) {
  int i = blockIdx.x * 256 + threadIdx.x;
  if (i < n4) {
    float4 v = *(const float4*)(in + (size_t)i * 4);
    unsigned short o[4] = { f2bf(v.x), f2bf(v.y), f2bf(v.z), f2bf(v.w) };
    *(uint2*)(out + (size_t)i * 4) = *(const uint2*)o;
  }
}

// ---------------------------------------------------------------------------
// Kernel 1b: fp32[K][N] -> bf16[N][K] convert + transpose (64x64 LDS tiles).
// Makes every weight tile a plain 2D copy for the TDM in the GEMMs.
// ---------------------------------------------------------------------------
__global__ __launch_bounds__(256) void cvt_tr_kernel(const float* __restrict__ in,
                                                     unsigned short* __restrict__ out,
                                                     int K, int N) {
  __shared__ unsigned short tile[64 * 68];  // padded pitch: no bank conflicts
  const int k0 = blockIdx.y * 64, n0 = blockIdx.x * 64;
  const int tid = threadIdx.x;
#pragma unroll
  for (int s = 0; s < 16; ++s) {
    const int idx = tid + s * 256;             // 4096 elements
    const int r = idx >> 6, c = idx & 63;
    tile[r * 68 + c] = f2bf(in[(size_t)(k0 + r) * N + n0 + c]);
  }
  __syncthreads();
#pragma unroll
  for (int s = 0; s < 16; ++s) {
    const int idx = tid + s * 256;
    const int nr = idx >> 6, kc = idx & 63;
    out[(size_t)(n0 + nr) * K + k0 + kc] = tile[kc * 68 + nr];
  }
}

// ---------------------------------------------------------------------------
// Kernel 2/4: tiled bf16 WMMA GEMM, C = A[8192,1024] * W^T (W stored [N][K]).
// Block tile 128x128, BK=32, 8 waves, wave tile 64x32. LDS double-buffered;
// tiles staged by TDM (wave 0 issues, S_WAIT_TENSORCNT 2 retires older stage).
// MODE 0: scatter bf16 into Q/K [B,H,T,dk] and V [B,H,dk,T] (+bias)
// MODE 1: fp32 [M, NTOT] output (+bias)  (the output projection)
// ---------------------------------------------------------------------------
template <int MODE, int NTOT>
__global__ __launch_bounds__(256) void gemm_bf16_kernel(
    const unsigned short* __restrict__ A, const unsigned short* __restrict__ W,
    const float* __restrict__ bias, void* __restrict__ outp) {
  __shared__ __attribute__((aligned(32))) unsigned short sA[2][128 * 32]; // [m][k]
  __shared__ __attribute__((aligned(32))) unsigned short sB[2][128 * 32]; // [n][k]

  const int tid = threadIdx.x;
  const int lane = tid & 31, w = tid >> 5;
  const int wm = (w & 1) * 64, wn = (w >> 1) * 32;
  const int mBase = blockIdx.y * 128, nBase = blockIdx.x * 128;

  v8f acc[4][2];
#pragma unroll
  for (int mt = 0; mt < 4; ++mt)
#pragma unroll
    for (int nt = 0; nt < 2; ++nt)
#pragma unroll
      for (int e = 0; e < 8; ++e) acc[mt][nt][e] = 0.0f;

  const int KT = CDIM / 32;
  // stage 0 in flight
  if (tid < 32) {
    tdm_load_2d(LDSOFF(&sA[0][0]), A + (size_t)mBase * CDIM, 32, 128, CDIM);
    tdm_load_2d(LDSOFF(&sB[0][0]), W + (size_t)nBase * CDIM, 32, 128, CDIM);
  }

  for (int kb = 0; kb < KT; ++kb) {
    const bool more = (kb + 1) < KT;
    if (tid < 32) {
      if (more) {
        const int k1 = (kb + 1) * 32, nb = (kb + 1) & 1;
        tdm_load_2d(LDSOFF(&sA[nb][0]), A + (size_t)mBase * CDIM + k1, 32, 128, CDIM);
        tdm_load_2d(LDSOFF(&sB[nb][0]), W + (size_t)nBase * CDIM + k1, 32, 128, CDIM);
        __builtin_amdgcn_s_wait_tensorcnt(2);   // retire stage kb, keep kb+1 flying
      } else {
        __builtin_amdgcn_s_wait_tensorcnt(0);
      }
    }
    __syncthreads();

    const unsigned short* a_lds = &sA[kb & 1][0];
    const unsigned short* b_lds = &sB[kb & 1][0];
    v16bf afrag[4], bfrag[2];
#pragma unroll
    for (int mt = 0; mt < 4; ++mt) afrag[mt] = load_frag_a(a_lds, 32, wm + mt * 16, 0, lane);
#pragma unroll
    for (int nt = 0; nt < 2; ++nt) bfrag[nt] = load_frag_b(b_lds, 32, wn + nt * 16, 0, lane);
#pragma unroll
    for (int mt = 0; mt < 4; ++mt)
#pragma unroll
      for (int nt = 0; nt < 2; ++nt)
        acc[mt][nt] = wmma_bf16(afrag[mt], bfrag[nt], acc[mt][nt]);
    __syncthreads();
  }

  // Epilogue. C-fragment element (vgpr r, lane): M = r + 8*(lane/16), N = lane%16
  const int ln = lane & 15, g = lane >> 4;
#pragma unroll
  for (int mt = 0; mt < 4; ++mt)
#pragma unroll
    for (int nt = 0; nt < 2; ++nt) {
      const int n = nBase + wn + nt * 16 + ln;
      const float bv = bias[n];
#pragma unroll
      for (int r = 0; r < 8; ++r) {
        const int m = mBase + wm + mt * 16 + r + 8 * g;
        const float v = acc[mt][nt][r] + bv;
        if (MODE == 1) {
          ((float*)outp)[(size_t)m * NTOT + n] = v;
        } else {
          const int which = n >> 10;      // 0=q 1=k 2=v
          const int c = n & 1023, h = c >> 6, d = c & 63;
          const int bb = m >> 11, t = m & 2047;
          size_t dst;
          if (which == 2)  // V stored transposed per head: [B,H,dk,T]
            dst = 2 * HEAD_ELEMS + (((size_t)bb * NH + h) * DK + d) * TSEQ + t;
          else
            dst = (size_t)which * HEAD_ELEMS + (((size_t)bb * NH + h) * TSEQ + t) * DK + d;
          ((unsigned short*)outp)[dst] = f2bf(v);
        }
      }
    }
}

// ---------------------------------------------------------------------------
// Kernel 3: flash attention (causal, online softmax), bf16 WMMA.
// grid = (T/128, B*H); block = 256 (8 waves). Wave w owns query rows
// [qBase + 16w, +16). Q/K/V tiles staged by TDM (V pre-transposed [dk][T]).
// Output y written bf16 as [B, T, C] (head-merged) for the projection GEMM.
// ---------------------------------------------------------------------------
__global__ __launch_bounds__(256) void flash_attn_kernel(
    const unsigned short* __restrict__ Qp, const unsigned short* __restrict__ Kp,
    const unsigned short* __restrict__ Vp, unsigned short* __restrict__ Y) {
  __shared__ __attribute__((aligned(32))) unsigned short sQ[128 * 64];  // [q][d]
  __shared__ __attribute__((aligned(32))) unsigned short sK[64 * 64];   // [kv][d]
  __shared__ __attribute__((aligned(32))) unsigned short sVt[64 * 64];  // [d][kv]
  __shared__ __attribute__((aligned(32))) unsigned short sP[8 * 16 * 64]; // per wave

  const int tid = threadIdx.x, lane = tid & 31, w = tid >> 5;
  const int ln = lane & 15, g = lane >> 4;
  const int bh = blockIdx.y;               // b*NH + h
  const int b = bh >> 4, h = bh & 15;
  const int qBase = blockIdx.x * 128;
  const size_t headOff = (size_t)bh * TSEQ * DK;   // same element count for Q/K/V

  // Stage the 128x64 Q block once (completion covered by first in-loop wait).
  if (tid < 32)
    tdm_load_2d(LDSOFF(sQ), Qp + headOff + (size_t)qBase * DK, 64, 128, 64);

  float mstat[8], lstat[8];
  v8f O[4];
#pragma unroll
  for (int r = 0; r < 8; ++r) { mstat[r] = -INFINITY; lstat[r] = 0.0f; }
#pragma unroll
  for (int t = 0; t < 4; ++t)
#pragma unroll
    for (int e = 0; e < 8; ++e) O[t][e] = 0.0f;

  unsigned short* myP = sP + w * 16 * 64;
  const int qRowMax = qBase + w * 16 + 15;
  const int nTiles = blockIdx.x * 2 + 2;   // kv tiles covering [0, qBase+128)

  for (int kt = 0; kt < nTiles; ++kt) {
    const int kvBase = kt * 64;
    __syncthreads();                       // previous round's reads of sK/sVt done
    if (tid < 32) {
      tdm_load_2d(LDSOFF(sK),  Kp + headOff + (size_t)kvBase * DK, 64, 64, 64);
      tdm_load_2d(LDSOFF(sVt), Vp + headOff + kvBase, 64, 64, TSEQ); // [dk][T]
      __builtin_amdgcn_s_wait_tensorcnt(0);
    }
    __syncthreads();

    if (kvBase > qRowMax) continue;  // wave-uniform: fully masked for this wave

    // ---- S = Q * K^T  (16 rows x 64 keys, f32 accum) ----
    v8f sfr[4];
#pragma unroll
    for (int n = 0; n < 4; ++n)
#pragma unroll
      for (int e = 0; e < 8; ++e) sfr[n][e] = 0.0f;
#pragma unroll
    for (int ks = 0; ks < 2; ++ks) {
      v16bf a = load_frag_a(sQ, 64, w * 16, ks * 32, lane);
#pragma unroll
      for (int n = 0; n < 4; ++n) {
        v16bf bk = load_frag_b(sK, 64, n * 16, ks * 32, lane);
        sfr[n] = wmma_bf16(a, bk, sfr[n]);
      }
    }

    // ---- scale, causal mask, per-row max ----
    float rmax[8];
#pragma unroll
    for (int r = 0; r < 8; ++r) rmax[r] = -INFINITY;
#pragma unroll
    for (int n = 0; n < 4; ++n) {
      const int kv = kvBase + n * 16 + ln;
#pragma unroll
      for (int r = 0; r < 8; ++r) {
        const int q = qBase + w * 16 + r + 8 * g;
        float v = sfr[n][r] * 0.125f;      // 1/sqrt(64)
        v = (kv <= q) ? v : -INFINITY;
        sfr[n][r] = v;
        rmax[r] = fmaxf(rmax[r], v);
      }
    }
#pragma unroll
    for (int r = 0; r < 8; ++r)
#pragma unroll
      for (int m = 1; m < 16; m <<= 1)
        rmax[r] = fmaxf(rmax[r], __shfl_xor(rmax[r], m, 32));

    float alpha[8], rsum[8];
#pragma unroll
    for (int r = 0; r < 8; ++r) {
      const float mn = fmaxf(mstat[r], rmax[r]);
      alpha[r] = __expf(mstat[r] - mn);
      mstat[r] = mn;
      rsum[r] = 0.0f;
    }

    // ---- P = exp(S - m), bf16 to per-wave LDS (A-operand staging) ----
#pragma unroll
    for (int n = 0; n < 4; ++n)
#pragma unroll
      for (int r = 0; r < 8; ++r) {
        const float p = __expf(sfr[n][r] - mstat[r]);
        rsum[r] += p;
        myP[(r + 8 * g) * 64 + n * 16 + ln] = f2bf(p);
      }
#pragma unroll
    for (int r = 0; r < 8; ++r) {
#pragma unroll
      for (int m = 1; m < 16; m <<= 1) rsum[r] += __shfl_xor(rsum[r], m, 32);
      lstat[r] = lstat[r] * alpha[r] + rsum[r];
    }

    // ---- O = O*alpha + P*V ----
#pragma unroll
    for (int t = 0; t < 4; ++t)
#pragma unroll
      for (int e = 0; e < 8; ++e) O[t][e] *= alpha[e];
#pragma unroll
    for (int ks = 0; ks < 2; ++ks) {
      v16bf a = load_frag_a(myP, 64, 0, ks * 32, lane);  // same-wave LDS: in-order
#pragma unroll
      for (int t = 0; t < 4; ++t) {
        v16bf bv = load_frag_b(sVt, 64, t * 16, ks * 32, lane);
        O[t] = wmma_bf16(a, bv, O[t]);
      }
    }
  }

  // ---- normalize and store y as [B, T, C] bf16 (c = h*64 + d) ----
#pragma unroll
  for (int t = 0; t < 4; ++t)
#pragma unroll
    for (int r = 0; r < 8; ++r) {
      const float v = O[t][r] / lstat[r];
      const int q = qBase + w * 16 + r + 8 * g;
      const int d = t * 16 + ln;
      Y[((size_t)(b * TSEQ + q) * NH + h) * DK + d] = f2bf(v);
    }
}

// ---------------------------------------------------------------------------
// Host-side launcher.  Workspace layout (bytes, all 256-aligned):
//   [0)           x_bf      16,777,216   [B*T, C] bf16
//   [16777216)    WqkvT_bf   6,291,456   [3C, C] bf16 (transposed)
//   [23068672)    WprojT_bf  2,097,152   [C, C]  bf16 (transposed)
//   [25165824)    Q|K|V bf  50,331,648   Q,K: [B,H,T,dk]; V: [B,H,dk,T]
//   [75497472)    y_bf      16,777,216   [B, T, C] bf16
// ---------------------------------------------------------------------------
extern "C" void kernel_launch(void* const* d_in, const int* in_sizes, int n_in,
                              void* d_out, int out_size, void* d_ws, size_t ws_size,
                              hipStream_t stream) {
  const float* x      = (const float*)d_in[0];
  const float* W_qkv  = (const float*)d_in[1];
  const float* b_qkv  = (const float*)d_in[2];
  const float* W_proj = (const float*)d_in[3];
  const float* b_proj = (const float*)d_in[4];

  char* ws = (char*)d_ws;
  unsigned short* x_bf   = (unsigned short*)(ws + 0);
  unsigned short* wqkvT  = (unsigned short*)(ws + 16777216);
  unsigned short* wprojT = (unsigned short*)(ws + 23068672);
  unsigned short* qkv    = (unsigned short*)(ws + 25165824);  // Q|K|V
  unsigned short* y_bf   = (unsigned short*)(ws + 75497472);

  // 1) fp32 -> bf16 conversions (weights transposed to [N][K])
  cvt_bf16_kernel<<<(MTOT * CDIM / 4) / 256, 256, 0, stream>>>(x, x_bf, MTOT * CDIM / 4);
  cvt_tr_kernel<<<dim3(3 * CDIM / 64, CDIM / 64), 256, 0, stream>>>(W_qkv, wqkvT, CDIM, 3 * CDIM);
  cvt_tr_kernel<<<dim3(CDIM / 64, CDIM / 64), 256, 0, stream>>>(W_proj, wprojT, CDIM, CDIM);

  // 2) QKV GEMM -> Q,K [B,H,T,dk] and V [B,H,dk,T]
  gemm_bf16_kernel<0, 3 * CDIM><<<dim3(3 * CDIM / 128, MTOT / 128), 256, 0, stream>>>(
      x_bf, wqkvT, b_qkv, (void*)qkv);

  // 3) causal flash attention -> y_bf [B, T, C]
  flash_attn_kernel<<<dim3(TSEQ / 128, BATCH * NH), 256, 0, stream>>>(
      qkv, qkv + HEAD_ELEMS, qkv + 2 * HEAD_ELEMS, y_bf);

  // 4) output projection: [8192,1024] x WprojT + bias -> fp32 d_out
  gemm_bf16_kernel<1, CDIM><<<dim3(CDIM / 128, MTOT / 128), 256, 0, stream>>>(
      y_bf, wprojT, b_proj, d_out);
}
